// CachedMultiHeadAttention_57930518888564
// MI455X (gfx1250) — matrix operand order repaired and checked by reading, not compiled, not run on hardware
//
#include <hip/hip_runtime.h>

// ---------------- problem constants ----------------
#define DMODEL 2048
#define NHEADS 16
#define DK     128
#define BATCH  2
#define SNEW   2048
#define PAST   2048
#define TOTAL  (PAST + SNEW)

typedef __bf16 bf16_t;
typedef __attribute__((ext_vector_type(16))) __bf16 v16bf;
typedef __attribute__((ext_vector_type(8)))  float  v8f;

union ABfrag { v16bf v; uint4 q[2]; };

__device__ inline v8f zero8() {
  v8f z = {0.f, 0.f, 0.f, 0.f, 0.f, 0.f, 0.f, 0.f};
  return z;
}

// A-matrix fragment (16x32 bf16), per CDNA5 ISA 7.12.2:
// lane L holds row m = L&15; VGPR0-3 hold K = 8*(L>>4)+0..7, VGPR4-7 hold K = 16+8*(L>>4)+0..7
__device__ inline v16bf ldfragA(const bf16_t* base, int stride, int k0, int lane) {
  const bf16_t* p = base + (size_t)(lane & 15) * stride + (k0 + 8 * (lane >> 4));
  ABfrag r;
  r.q[0] = *(const uint4*)(p);
  r.q[1] = *(const uint4*)(p + 16);
  return r.v;
}

// B-matrix fragment (32x16 bf16): lane L holds column n = L&15;
// K = 8*(L>>4)+0..7 then 16+8*(L>>4)+0..7 (same order as A)
__device__ inline v16bf ldfragB(const bf16_t* base, int stride, int k0, int lane) {
  const bf16_t* p = base + (size_t)(lane & 15) * stride + (k0 + 8 * (lane >> 4));
  ABfrag r;
  r.q[0] = *(const uint4*)(p);
  r.q[1] = *(const uint4*)(p + 16);
  return r.v;
}

#define WMMA_BF16(a, b, c) \
  __builtin_amdgcn_wmma_f32_16x16x32_bf16(false, (a), false, (b), (short)0, (c), false, false)

// xor-swizzle across lanes; masks < 16 stay within each 16-lane half (matches C-frag row layout)
#define SWZ_F(x, m) \
  __int_as_float(__builtin_amdgcn_ds_swizzle(__float_as_int(x), (0x1f | ((m) << 10))))

__device__ inline void rowmax16(float v[8]) {
#pragma unroll
  for (int r = 0; r < 8; r++) {
    float t = v[r];
    t = fmaxf(t, SWZ_F(t, 1));
    t = fmaxf(t, SWZ_F(t, 2));
    t = fmaxf(t, SWZ_F(t, 4));
    t = fmaxf(t, SWZ_F(t, 8));
    v[r] = t;
  }
}

__device__ inline void rowsum16(float v[8]) {
#pragma unroll
  for (int r = 0; r < 8; r++) {
    float t = v[r];
    t += SWZ_F(t, 1);
    t += SWZ_F(t, 2);
    t += SWZ_F(t, 4);
    t += SWZ_F(t, 8);
    v[r] = t;
  }
}

// ---------------- gfx1250 async global->LDS copy (ASYNCcnt tracked) -----------------------------
__device__ inline void async_copy16(unsigned lds_off, const bf16_t* gptr) {
  asm volatile("global_load_async_to_lds_b128 %0, %1, off"
               :: "v"(lds_off), "v"((unsigned long long)(uintptr_t)gptr)
               : "memory");
}
__device__ inline void wait_async_zero() {
  asm volatile("s_wait_asynccnt 0x0" ::: "memory");
}

// ---------------- shared GEMM mainloop: C[m,n] = sum_k A[m,k]*B[n,k], bf16 operands -------------
#define LDT 72  // LDS row stride in bf16 (64 + 8 pad; keeps 16B alignment, dodges bank conflicts)

// stage one 128x64 bf16 tile of A and of B into LDS via async loads (8 ops/thread, 16B each)
__device__ inline void stage_tiles_async(const bf16_t* Ag, const bf16_t* Bg,
                                         bf16_t* As, bf16_t* Bs,
                                         int m_base, int n_base, int k0, int tid) {
#pragma unroll
  for (int i = 0; i < 4; i++) {
    const int c = tid + i * 256;      // 1024 16B chunks per matrix
    const int row = c >> 3;           // 0..127
    const int sub = (c & 7) * 8;      // bf16 col offset 0..56
    async_copy16((unsigned)(uintptr_t)&As[row * LDT + sub],
                 Ag + (size_t)(m_base + row) * DMODEL + k0 + sub);
    async_copy16((unsigned)(uintptr_t)&Bs[row * LDT + sub],
                 Bg + (size_t)(n_base + row) * DMODEL + k0 + sub);
  }
}

__device__ inline void gemm_tile(const bf16_t* Ag, const bf16_t* Bg,
                                 int m_base, int n_base, v8f acc[4][2]) {
  __shared__ bf16_t As[2][128 * LDT];
  __shared__ bf16_t Bs[2][128 * LDT];
  const int tid  = threadIdx.x;
  const int lane = tid & 31;
  const int w    = tid >> 5;
  const int wm   = (w >> 2) * 64;  // 2 waves in M
  const int wn   = (w & 3) * 32;   // 4 waves in N

#pragma unroll
  for (int i = 0; i < 4; i++) { acc[i][0] = zero8(); acc[i][1] = zero8(); }

  stage_tiles_async(Ag, Bg, As[0], Bs[0], m_base, n_base, 0, tid);

  for (int k0 = 0; k0 < DMODEL; k0 += 64) {
    const int buf = (k0 >> 6) & 1;
    wait_async_zero();      // tile k resident (per-wave)
    __syncthreads();        // all waves: tile k resident, tile k-1 compute done
    if (k0 + 64 < DMODEL)   // stream tile k+1 into the other buffer while computing tile k
      stage_tiles_async(Ag, Bg, As[buf ^ 1], Bs[buf ^ 1], m_base, n_base, k0 + 64, tid);
#pragma unroll
    for (int ks = 0; ks < 64; ks += 32) {
      // batch all fragment loads so the ds_loads pipeline behind one wait,
      // then issue the 8 WMMAs back-to-back
      v16bf b0 = ldfragB(&Bs[buf][(wn + 0)  * LDT], LDT, ks, lane);
      v16bf b1 = ldfragB(&Bs[buf][(wn + 16) * LDT], LDT, ks, lane);
      v16bf a0 = ldfragA(&As[buf][(wm + 0)  * LDT], LDT, ks, lane);
      v16bf a1 = ldfragA(&As[buf][(wm + 16) * LDT], LDT, ks, lane);
      v16bf a2 = ldfragA(&As[buf][(wm + 32) * LDT], LDT, ks, lane);
      v16bf a3 = ldfragA(&As[buf][(wm + 48) * LDT], LDT, ks, lane);
      acc[0][0] = WMMA_BF16(a0, b0, acc[0][0]);
      acc[0][1] = WMMA_BF16(a0, b1, acc[0][1]);
      acc[1][0] = WMMA_BF16(a1, b0, acc[1][0]);
      acc[1][1] = WMMA_BF16(a1, b1, acc[1][1]);
      acc[2][0] = WMMA_BF16(a2, b0, acc[2][0]);
      acc[2][1] = WMMA_BF16(a2, b1, acc[2][1]);
      acc[3][0] = WMMA_BF16(a3, b0, acc[3][0]);
      acc[3][1] = WMMA_BF16(a3, b1, acc[3][1]);
    }
  }
}

// ---------------- kernel 0: bulk f32 -> bf16 conversion (x, weights) ----------------------------
__global__ void __launch_bounds__(256) k_cvt_f32_bf16(const float* __restrict__ src,
                                                      bf16_t* __restrict__ dst) {
  const size_t i = ((size_t)blockIdx.x * 256 + threadIdx.x) * 8;
  float4 a = *(const float4*)(src + i);
  float4 b = *(const float4*)(src + i + 4);
  bf16_t t[8] = {(bf16_t)a.x, (bf16_t)a.y, (bf16_t)a.z, (bf16_t)a.w,
                 (bf16_t)b.x, (bf16_t)b.y, (bf16_t)b.z, (bf16_t)b.w};
  *(uint4*)(dst + i) = *(uint4*)t;
}

// ---------------- kernel 1a: past_key f32 -> bf16 (same layout, into rows [0,PAST)) -------------
__global__ void __launch_bounds__(256) k_cvt_past_key(const float* __restrict__ pk,
                                                      bf16_t* __restrict__ Kbf) {
  const size_t i   = ((size_t)blockIdx.x * 256 + threadIdx.x) * 4;
  float4 f = *(const float4*)(pk + i);
  const size_t bh  = i / ((size_t)PAST * DK);
  const size_t rem = i - bh * (size_t)PAST * DK;
  bf16_t* dst = Kbf + bh * (size_t)TOTAL * DK + rem;
  bf16_t t[4] = {(bf16_t)f.x, (bf16_t)f.y, (bf16_t)f.z, (bf16_t)f.w};
  *(uint2*)dst = *(uint2*)t;
}

// ---------------- kernel 1b: past_value f32 [bh][p][d] -> bf16 transposed Vt[bh][d][p] ----------
__global__ void __launch_bounds__(256) k_cvt_past_value(const float* __restrict__ pv,
                                                        bf16_t* __restrict__ Vt) {
  __shared__ float tile[32][33];
  const int bh = blockIdx.z;
  const int p0 = blockIdx.x * 32;
  const int d0 = blockIdx.y * 32;
  const int tx = threadIdx.x & 31;
  const int ty = threadIdx.x >> 5;  // 0..7
#pragma unroll
  for (int i = 0; i < 4; i++) {
    const int p = p0 + ty + i * 8;
    tile[ty + i * 8][tx] = pv[((size_t)bh * PAST + p) * DK + d0 + tx];
  }
  __syncthreads();
#pragma unroll
  for (int i = 0; i < 4; i++) {
    const int d = d0 + ty + i * 8;
    Vt[((size_t)bh * DK + d) * TOTAL + p0 + tx] = (bf16_t)tile[tx][ty + i * 8];
  }
}

// ---------------- kernel 2: fused QKV projection (y = x @ W^T), bf16 head-split outputs ---------
__global__ void __launch_bounds__(256) k_qkv_gemm(const bf16_t* __restrict__ xbf,
                                                  const bf16_t* __restrict__ Wqb,
                                                  const bf16_t* __restrict__ Wkb,
                                                  const bf16_t* __restrict__ Wvb,
                                                  bf16_t* __restrict__ Qbf,
                                                  bf16_t* __restrict__ Kbf,
                                                  bf16_t* __restrict__ Vt) {
  const int z = blockIdx.z;
  const bf16_t* W = (z == 0) ? Wqb : ((z == 1) ? Wkb : Wvb);
  const int m_base = blockIdx.y * 128;
  const int n_base = blockIdx.x * 128;
  v8f acc[4][2];
  gemm_tile(xbf, W, m_base, n_base, acc);

  const int tid = threadIdx.x, lane = tid & 31, w = tid >> 5;
  const int wm = m_base + (w >> 2) * 64;
  const int wn = n_base + (w & 3) * 32;
#pragma unroll
  for (int mf = 0; mf < 4; mf++)
#pragma unroll
    for (int nf = 0; nf < 2; nf++)
#pragma unroll
      for (int r = 0; r < 8; r++) {
        const int m = wm + mf * 16 + (lane >> 4) * 8 + r;
        const int n = wn + nf * 16 + (lane & 15);
        const int b = m >> 11, s = m & (SNEW - 1);
        const int h = n >> 7,  d = n & (DK - 1);
        const bf16_t val = (bf16_t)acc[mf][nf][r];
        if (z == 0)
          Qbf[((size_t)(b * NHEADS + h) * SNEW + s) * DK + d] = val;
        else if (z == 1)
          Kbf[((size_t)(b * NHEADS + h) * TOTAL + PAST + s) * DK + d] = val;
        else
          Vt[((size_t)(b * NHEADS + h) * DK + d) * TOTAL + PAST + s] = val;
      }
}

// ---------------- kernel 3: causal flash attention, bf16 WMMA, f32 online softmax ---------------
__global__ void __launch_bounds__(256) k_attention(const bf16_t* __restrict__ Qbf,
                                                   const bf16_t* __restrict__ Kbf,
                                                   const bf16_t* __restrict__ Vt,
                                                   bf16_t* __restrict__ AO) {
  __shared__ bf16_t Plds[8][16 * 40];  // per-wave 16x32 P staging (stride 40 -> 16B aligned)
  const int bh = blockIdx.y;
  const int b = bh >> 4, h = bh & 15;
  const int tid = threadIdx.x, lane = tid & 31, w = tid >> 5;
  const int q0 = blockIdx.x * 128 + w * 16;  // this wave's 16 query rows

  const bf16_t* Qb = Qbf + ((size_t)bh * SNEW + q0) * DK;
  const bf16_t* Kb = Kbf + (size_t)bh * TOTAL * DK;
  const bf16_t* Vb = Vt  + (size_t)bh * DK * TOTAL;
  bf16_t* pw = &Plds[w][0];

  v16bf qf[4];
#pragma unroll
  for (int kd = 0; kd < 4; kd++) qf[kd] = ldfragA(Qb, DK, kd * 32, lane);

  v8f o[8];
#pragma unroll
  for (int i = 0; i < 8; i++) o[i] = zero8();
  float rm[8], rl[8];
#pragma unroll
  for (int r = 0; r < 8; r++) { rm[r] = -1e30f; rl[r] = 0.f; }

  const float sc2 = 0.08838834764831845f * 1.4426950408889634f;  // (1/sqrt(dk)) * log2(e)
  const int kv_end = PAST + q0 + 16;  // causal limit for this wave

  for (int kt = 0; kt < kv_end; kt += 32) {
    // prefetch next K/V tile into cache while computing this one
    if (kt + 32 < kv_end) {
      __builtin_prefetch(Kb + (size_t)(kt + 32 + lane) * DK, 0, 1);
      __builtin_prefetch(Vb + (size_t)(lane * 4) * TOTAL + kt + 32, 0, 1);
    }
    const bool need_mask = (kt + 31) > (PAST + q0);
    // ---- S = Q K^T (16 x 32): batch the 8 global fragment loads, then 8 WMMAs ----
    v16bf kb[2][4];
#pragma unroll
    for (int nf = 0; nf < 2; nf++)
#pragma unroll
      for (int kd = 0; kd < 4; kd++)
        kb[nf][kd] = ldfragB(Kb + (size_t)(kt + nf * 16) * DK, DK, kd * 32, lane);
    float sv[2][8];
#pragma unroll
    for (int nf = 0; nf < 2; nf++) {
      v8f s = zero8();
#pragma unroll
      for (int kd = 0; kd < 4; kd++) s = WMMA_BF16(qf[kd], kb[nf][kd], s);
#pragma unroll
      for (int r = 0; r < 8; r++) {
        float vsc = s[r] * sc2;
        if (need_mask) {
          const int kv = kt + nf * 16 + (lane & 15);
          const int qa = PAST + q0 + (lane >> 4) * 8 + r;
          if (kv > qa) vsc = -1e30f;
        }
        sv[nf][r] = vsc;
      }
    }
    // ---- online softmax (base-2 domain) ----
    float tmax[8];
#pragma unroll
    for (int r = 0; r < 8; r++) tmax[r] = fmaxf(sv[0][r], sv[1][r]);
    rowmax16(tmax);
    float alpha[8];
#pragma unroll
    for (int r = 0; r < 8; r++) {
      const float mn = fmaxf(rm[r], tmax[r]);
      alpha[r] = exp2f(rm[r] - mn);
      rm[r] = mn;
    }
    float ps[8];
#pragma unroll
    for (int r = 0; r < 8; r++) ps[r] = 0.f;
#pragma unroll
    for (int nf = 0; nf < 2; nf++)
#pragma unroll
      for (int r = 0; r < 8; r++) {
        const float p = exp2f(sv[nf][r] - rm[r]);
        ps[r] += p;
        // C-layout (row=(lane>>4)*8+r, col=nf*16+(lane&15)) -> LDS row-major
        pw[((lane >> 4) * 8 + r) * 40 + nf * 16 + (lane & 15)] = (bf16_t)p;
      }
    rowsum16(ps);
#pragma unroll
    for (int r = 0; r < 8; r++) rl[r] = rl[r] * alpha[r] + ps[r];
#pragma unroll
    for (int i = 0; i < 8; i++)
#pragma unroll
      for (int r = 0; r < 8; r++) o[i][r] *= alpha[r];
    // ---- O += P * V  (read P back as A-fragment; Vt gives K-contiguous B-fragments) ----
    v16bf pa = ldfragA(pw, 40, 0, lane);
#pragma unroll
    for (int half = 0; half < 2; half++) {
      // groups of 4: enough load-level parallelism without blowing VGPR pressure
      v16bf vb[4];
#pragma unroll
      for (int j = 0; j < 4; j++)
        vb[j] = ldfragB(Vb + (size_t)((half * 4 + j) * 16) * TOTAL, TOTAL, kt, lane);
#pragma unroll
      for (int j = 0; j < 4; j++)
        o[half * 4 + j] = WMMA_BF16(pa, vb[j], o[half * 4 + j]);
    }
  }
  // ---- normalize + write attn-out bf16 [b][s][h*128+d] ----
#pragma unroll
  for (int r = 0; r < 8; r++) rl[r] = 1.f / rl[r];
#pragma unroll
  for (int nf2 = 0; nf2 < 8; nf2++)
#pragma unroll
    for (int r = 0; r < 8; r++) {
      const int srow = q0 + (lane >> 4) * 8 + r;
      const int d = nf2 * 16 + (lane & 15);
      AO[((size_t)b * SNEW + srow) * DMODEL + h * DK + d] = (bf16_t)(o[nf2][r] * rl[r]);
    }
}

// ---------------- kernel 4: output projection (out = AO @ Wo^T), f32 output ---------------------
__global__ void __launch_bounds__(256) k_out_gemm(const bf16_t* __restrict__ AO,
                                                  const bf16_t* __restrict__ Wob,
                                                  float* __restrict__ out) {
  const int m_base = blockIdx.y * 128;
  const int n_base = blockIdx.x * 128;
  v8f acc[4][2];
  gemm_tile(AO, Wob, m_base, n_base, acc);

  const int tid = threadIdx.x, lane = tid & 31, w = tid >> 5;
  const int wm = m_base + (w >> 2) * 64;
  const int wn = n_base + (w & 3) * 32;
#pragma unroll
  for (int mf = 0; mf < 4; mf++)
#pragma unroll
    for (int nf = 0; nf < 2; nf++)
#pragma unroll
      for (int r = 0; r < 8; r++) {
        const int m = wm + mf * 16 + (lane >> 4) * 8 + r;
        const int n = wn + nf * 16 + (lane & 15);
        out[(size_t)m * DMODEL + n] = acc[mf][nf][r];
      }
}

// ---------------- launcher ----------------------------------------------------------------------
extern "C" void kernel_launch(void* const* d_in, const int* in_sizes, int n_in,
                              void* d_out, int out_size, void* d_ws, size_t ws_size,
                              hipStream_t stream) {
  const float* x  = (const float*)d_in[0];
  const float* pk = (const float*)d_in[1];
  const float* pv = (const float*)d_in[2];
  const float* Wq = (const float*)d_in[3];
  const float* Wk = (const float*)d_in[4];
  const float* Wv = (const float*)d_in[5];
  const float* Wo = (const float*)d_in[6];
  float* out = (float*)d_out;

  // workspace layout (bf16): Q | K(full) | Vt(full,transposed) | attn_out | x | Wq Wk Wv Wo
  // = (3*8.4M + 2*16.8M + 4*4.2M) elems * 2B  ~= 144 MB
  const size_t nQ  = (size_t)BATCH * NHEADS * SNEW * DK;    // 8,388,608
  const size_t nKV = (size_t)BATCH * NHEADS * TOTAL * DK;   // 16,777,216
  const size_t nX  = (size_t)BATCH * SNEW * DMODEL;         // 8,388,608
  const size_t nW  = (size_t)DMODEL * DMODEL;               // 4,194,304
  bf16_t* Qbf = (bf16_t*)d_ws;
  bf16_t* Kbf = Qbf + nQ;
  bf16_t* Vt  = Kbf + nKV;
  bf16_t* AO  = Vt + nKV;
  bf16_t* Xbf = AO + nX;
  bf16_t* Wqb = Xbf + nX;
  bf16_t* Wkb = Wqb + nW;
  bf16_t* Wvb = Wkb + nW;
  bf16_t* Wob = Wvb + nW;

  // 0) one-time f32 -> bf16 downcasts (x + 4 weights)
  k_cvt_f32_bf16<<<dim3((unsigned)(nX / 2048)), 256, 0, stream>>>(x, Xbf);
  k_cvt_f32_bf16<<<dim3((unsigned)(nW / 2048)), 256, 0, stream>>>(Wq, Wqb);
  k_cvt_f32_bf16<<<dim3((unsigned)(nW / 2048)), 256, 0, stream>>>(Wk, Wkb);
  k_cvt_f32_bf16<<<dim3((unsigned)(nW / 2048)), 256, 0, stream>>>(Wv, Wvb);
  k_cvt_f32_bf16<<<dim3((unsigned)(nW / 2048)), 256, 0, stream>>>(Wo, Wob);
  // 1) convert/transpose the KV cache
  k_cvt_past_key<<<dim3((unsigned)(((size_t)BATCH * NHEADS * PAST * DK) / 1024)), 256, 0, stream>>>(pk, Kbf);
  k_cvt_past_value<<<dim3(PAST / 32, DK / 32, BATCH * NHEADS), 256, 0, stream>>>(pv, Vt);
  // 2) Q/K/V projections (z picks the weight + destination)
  k_qkv_gemm<<<dim3(DMODEL / 128, (BATCH * SNEW) / 128, 3), 256, 0, stream>>>(Xbf, Wqb, Wkb, Wvb, Qbf, Kbf, Vt);
  // 3) causal flash attention
  k_attention<<<dim3(SNEW / 128, BATCH * NHEADS), 256, 0, stream>>>(Qbf, Kbf, Vt, AO);
  // 4) output projection
  k_out_gemm<<<dim3(DMODEL / 128, (BATCH * SNEW) / 128), 256, 0, stream>>>(AO, Wob, out);
}
